// GCN_4612794876643
// MI455X (gfx1250) — matrix operand order repaired and checked by reading, not compiled, not run on hardware
//
#include <hip/hip_runtime.h>
#include <hip/hip_bf16.h>

typedef __attribute__((ext_vector_type(16))) __bf16 v16bf;
typedef __attribute__((ext_vector_type(8)))  __bf16 v8bf;
typedef __attribute__((ext_vector_type(8)))  float  v8f;

#define N_NODES 100000
#define FEAT0 256
#define NHID  128
#define NCLS  40
#define ROWS_PB 64          // rows per block (4 x 16 sub-tiles)

// ---------------------------------------------------------------------------
// Pack a row-major fp32 weight [K x N] into bf16 fragment-order:
//   Bp[((k/16)*Npad + n)*16 + (k%16)]   (n >= N zero-padded)
// so a WMMA lane's B fragment (16 consecutive K for one n) is contiguous.
// ---------------------------------------------------------------------------
__global__ void pack_w_kernel(const float* __restrict__ W, __bf16* __restrict__ Bp,
                              int K, int N, int Npad) {
    int idx = blockIdx.x * blockDim.x + threadIdx.x;
    int total = K * Npad;
    if (idx >= total) return;
    int t = idx & 15;              // k % 16
    int n = (idx >> 4) % Npad;
    int c = idx / (16 * Npad);     // k / 16
    int k = c * 16 + t;
    float v = (n < N) ? W[(size_t)k * N + n] : 0.0f;
    Bp[idx] = (__bf16)v;
}

__global__ void fill_f32_kernel(float* __restrict__ p, float v, long long n4) {
    long long i = (long long)blockIdx.x * blockDim.x + threadIdx.x;
    if (i >= n4) return;
    float4 x = make_float4(v, v, v, v);
    ((float4*)p)[i] = x;
}

__global__ void degree_kernel(const long long* __restrict__ dst, float* __restrict__ deg,
                              long long nE) {
    long long e = (long long)blockIdx.x * blockDim.x + threadIdx.x;
    if (e >= nE) return;
    atomicAdd(&deg[(int)dst[e]], 1.0f);
}

__global__ void rsqrt_kernel(float* __restrict__ deg, int n) {
    int i = blockIdx.x * blockDim.x + threadIdx.x;
    if (i >= n) return;
    deg[i] = rsqrtf(deg[i] + 1.0f);   // deg includes self loop
}

// ---------------------------------------------------------------------------
// C[M x N] = A[M x K] @ W  using v_wmma_f32_16x16x32_bf16.
// Block = (Npad/16) waves, computes a 64-row x Npad tile.
// A tile staged as bf16 in LDS (coalesced fp32 load + convert).
// Each wave holds 4 accumulators (4 row sub-tiles) and reuses one B
// fragment (2 x global b128) across 4 WMMAs -> 2.5 mem-ops per WMMA.
// ---------------------------------------------------------------------------
__global__ void gemm_wmma_kernel(const float* __restrict__ A,
                                 const __bf16* __restrict__ Bp,
                                 const float* __restrict__ bias,   // may be null
                                 float* __restrict__ C,
                                 int M, int K, int N, int Npad) {
    __shared__ __bf16 ldsA[ROWS_PB * 256];     // up to K = 256 -> 32 KB

    const int row0 = blockIdx.x * ROWS_PB;

    // ---- cooperative load ROWS_PB x K fp32 -> bf16 LDS (rows >= M -> 0) ----
    const int nChunks = (ROWS_PB * K) >> 2;            // float4 chunks
    for (int i = threadIdx.x; i < nChunks; i += blockDim.x) {
        int elem = i << 2;
        int r = elem / K;
        int cidx = elem - r * K;
        float4 v = make_float4(0.f, 0.f, 0.f, 0.f);
        int row = row0 + r;
        if (row < M) v = *(const float4*)(A + (size_t)row * K + cidx);
        __bf16* d = &ldsA[r * K + cidx];
        d[0] = (__bf16)v.x; d[1] = (__bf16)v.y; d[2] = (__bf16)v.z; d[3] = (__bf16)v.w;
    }
    __syncthreads();

    const int wave   = threadIdx.x >> 5;
    const int lane   = threadIdx.x & 31;
    const int half   = lane >> 4;         // K-half select: k = 16*half + i (contiguous order)
    const int m_loc  = lane & 15;         // A fragment row within sub-tile
    const int n_loc  = lane & 15;         // B/C fragment column
    const int ncol0  = wave * 16;

    v8f acc0 = {}, acc1 = {}, acc2 = {}, acc3 = {};

    const int ksteps = K >> 5;            // K / 32
    for (int s = 0; s < ksteps; ++s) {
        const int k0 = s << 5;

        // B fragment (shared by all 4 row sub-tiles):
        // packed chunk c = s*2 + half, column ncol0+n_loc, 16 contiguous bf16
        const v8bf* bp = (const v8bf*)&Bp[(((size_t)(s * 2 + half) * Npad) + (ncol0 + n_loc)) * 16];
        v8bf b0 = bp[0];
        v8bf b1 = bp[1];
        v16bf bv;
        #pragma unroll
        for (int i = 0; i < 8; ++i) { bv[i] = b0[i]; bv[8 + i] = b1[i]; }

        #pragma unroll
        for (int sub = 0; sub < 4; ++sub) {
            const int mrow = sub * 16 + m_loc;
            v8bf a0 = *(const v8bf*)&ldsA[mrow * K + k0 + half * 16];
            v8bf a1 = *(const v8bf*)&ldsA[mrow * K + k0 + half * 16 + 8];
            v16bf av;
            #pragma unroll
            for (int i = 0; i < 8; ++i) { av[i] = a0[i]; av[8 + i] = a1[i]; }
            v8f& acc = (sub == 0) ? acc0 : (sub == 1) ? acc1 : (sub == 2) ? acc2 : acc3;
            acc = __builtin_amdgcn_wmma_f32_16x16x32_bf16(
                /*neg_a=*/false, av, /*neg_b=*/false, bv,
                /*c_mod=*/(short)0, acc, /*reuse_a=*/false, /*reuse_b=*/false);
        }
    }

    // ---- epilogue: lane n = lane&15, VGPR r holds m = r + 8*half ----
    int n = ncol0 + n_loc;
    if (n < N) {
        float bval = (bias != nullptr) ? bias[n] : 0.0f;
        #pragma unroll
        for (int sub = 0; sub < 4; ++sub) {
            const v8f& acc = (sub == 0) ? acc0 : (sub == 1) ? acc1 : (sub == 2) ? acc2 : acc3;
            #pragma unroll
            for (int r = 0; r < 8; ++r) {
                int row = row0 + sub * 16 + r + 8 * half;
                if (row < M) C[(size_t)row * N + n] = acc[r] + bval;
            }
        }
    }
}

// ---------------------------------------------------------------------------
// One wave32 per edge: agg[dst] += h[src] * dinv[src]*dinv[dst]  (128 cols)
// ---------------------------------------------------------------------------
__global__ void aggregate_kernel(const float* __restrict__ h, float* __restrict__ agg,
                                 const long long* __restrict__ src,
                                 const long long* __restrict__ dst,
                                 const float* __restrict__ dinv, long long nE) {
    int lane = threadIdx.x & 31;
    long long e = (long long)blockIdx.x * (blockDim.x >> 5) + (threadIdx.x >> 5);
    if (e >= nE) return;
    int s = (int)src[e];
    int d = (int)dst[e];
    float w = dinv[s] * dinv[d];
    float4 v = ((const float4*)(h + (size_t)s * NHID))[lane];
    float* ad = agg + (size_t)d * NHID + lane * 4;
    atomicAdd(&ad[0], v.x * w);
    atomicAdd(&ad[1], v.y * w);
    atomicAdd(&ad[2], v.z * w);
    atomicAdd(&ad[3], v.w * w);
}

// agg = relu(agg + h * dinv^2 + bias)  -- one thread per (node, 4 cols)
__global__ void finalize_kernel(float* __restrict__ agg, const float* __restrict__ h,
                                const float* __restrict__ dinv,
                                const float* __restrict__ bias, int M) {
    long long idx = (long long)blockIdx.x * blockDim.x + threadIdx.x;
    if (idx >= (long long)M * 32) return;
    int node = (int)(idx >> 5);
    int q = (int)(idx & 31);
    float di = dinv[node];
    float sl = di * di;
    size_t off = (size_t)node * NHID + q * 4;
    float4 a = *(float4*)(agg + off);
    float4 hv = *(const float4*)(h + off);
    a.x = fmaxf(a.x + hv.x * sl + bias[q * 4 + 0], 0.0f);
    a.y = fmaxf(a.y + hv.y * sl + bias[q * 4 + 1], 0.0f);
    a.z = fmaxf(a.z + hv.z * sl + bias[q * 4 + 2], 0.0f);
    a.w = fmaxf(a.w + hv.w * sl + bias[q * 4 + 3], 0.0f);
    *(float4*)(agg + off) = a;
}

extern "C" void kernel_launch(void* const* d_in, const int* in_sizes, int n_in,
                              void* d_out, int out_size, void* d_ws, size_t ws_size,
                              hipStream_t stream) {
    const float*      x    = (const float*)d_in[0];
    const long long*  ei   = (const long long*)d_in[1];   // int64 [2, nE]
    const float*      W1   = (const float*)d_in[2];
    const float*      b1   = (const float*)d_in[3];
    const float*      W2   = (const float*)d_in[4];
    const float*      b2   = (const float*)d_in[5];
    const float*      Wc   = (const float*)d_in[6];
    const float*      bc   = (const float*)d_in[7];
    float*            out  = (float*)d_out;

    const long long nE = (long long)in_sizes[1] / 2;
    const long long* src = ei;
    const long long* dst = ei + nE;
    const int M = N_NODES;

    // ---- workspace carve-up (256 B aligned) ----
    char* ws = (char*)d_ws;
    size_t off = 0;
    auto carve = [&](size_t bytes) -> void* {
        void* p = ws + off;
        off = (off + bytes + 255) & ~(size_t)255;
        return p;
    };
    float*  dinv = (float*)carve((size_t)M * sizeof(float));
    float*  bufA = (float*)carve((size_t)M * NHID * sizeof(float));
    float*  bufB = (float*)carve((size_t)M * NHID * sizeof(float));
    __bf16* W1p  = (__bf16*)carve((size_t)FEAT0 * NHID * sizeof(__bf16));
    __bf16* W2p  = (__bf16*)carve((size_t)NHID * NHID * sizeof(__bf16));
    __bf16* Wcp  = (__bf16*)carve((size_t)NHID * 48 * sizeof(__bf16));
    (void)ws_size; (void)n_in; (void)out_size;

    const int TB = 256;

    // 1) pack weights into bf16 fragment order
    pack_w_kernel<<<(FEAT0 * NHID + TB - 1) / TB, TB, 0, stream>>>(W1, W1p, FEAT0, NHID, NHID);
    pack_w_kernel<<<(NHID * NHID + TB - 1) / TB, TB, 0, stream>>>(W2, W2p, NHID, NHID, NHID);
    pack_w_kernel<<<(NHID * 48 + TB - 1) / TB, TB, 0, stream>>>(Wc, Wcp, NHID, NCLS, 48);

    // 2) degrees -> dinv
    fill_f32_kernel<<<(int)((M / 4 + TB - 1) / TB), TB, 0, stream>>>(dinv, 0.0f, M / 4);
    degree_kernel<<<(int)((nE + TB - 1) / TB), TB, 0, stream>>>(dst, dinv, nE);
    rsqrt_kernel<<<(M + TB - 1) / TB, TB, 0, stream>>>(dinv, M);

    const int rowTiles = (M + ROWS_PB - 1) / ROWS_PB;

    // 3) layer 1: h1 = x @ W1 (bf16 WMMA, f32 accum)
    gemm_wmma_kernel<<<rowTiles, (NHID / 16) * 32, 0, stream>>>(x, W1p, nullptr, bufA,
                                                                M, FEAT0, NHID, NHID);
    //    aggregate + self-loop + bias + relu
    fill_f32_kernel<<<(int)(((long long)M * NHID / 4 + TB - 1) / TB), TB, 0, stream>>>(
        bufB, 0.0f, (long long)M * NHID / 4);
    aggregate_kernel<<<(int)((nE + 7) / 8), TB, 0, stream>>>(bufA, bufB, src, dst, dinv, nE);
    finalize_kernel<<<(int)(((long long)M * 32 + TB - 1) / TB), TB, 0, stream>>>(
        bufB, bufA, dinv, b1, M);

    // 4) layer 2: h2 = r1 @ W2
    gemm_wmma_kernel<<<rowTiles, (NHID / 16) * 32, 0, stream>>>(bufB, W2p, nullptr, bufA,
                                                                M, NHID, NHID, NHID);
    fill_f32_kernel<<<(int)(((long long)M * NHID / 4 + TB - 1) / TB), TB, 0, stream>>>(
        bufB, 0.0f, (long long)M * NHID / 4);
    aggregate_kernel<<<(int)((nE + 7) / 8), TB, 0, stream>>>(bufA, bufB, src, dst, dinv, nE);
    finalize_kernel<<<(int)(((long long)M * 32 + TB - 1) / TB), TB, 0, stream>>>(
        bufB, bufA, dinv, b2, M);

    // 5) classifier: out = r2 @ Wc + bc   (Npad = 48 -> 3 waves/block)
    gemm_wmma_kernel<<<rowTiles, (48 / 16) * 32, 0, stream>>>(bufB, Wcp, bc, out,
                                                              M, NHID, NCLS, 48);
}
